// Module_43508018708785
// MI455X (gfx1250) — matrix operand order repaired and checked by reading, not compiled, not run on hardware
//
#include <hip/hip_runtime.h>
#include <hip/hip_bf16.h>
#include <math.h>

#define N_USERS 100000
#define N_ITEMS 50000
#define DIMD    64
#define HLEN    100
#define BATCH   4096
#define LN_EPS  1e-5f

#define HP      112           // history padded to 7*16 rows
#define NTILE   7
#define H1_LD   136           // 128 cols + 8 halves pad (stride 272B -> conflict-free rows)

#define SW1_H   32768         // 256x128 f16, fragment order
#define SW2_H   8192          // 128x64  f16, fragment order
#define SUE_H   128
#define SH1_H   (NTILE * 16 * H1_LD)
#define NFLOATS (128*3 + 64*4 + 112 + 8 + 256 + 256)
#define SMEM_BYTES ((SW1_H + SW2_H + SUE_H + SH1_H) * 2 + NFLOATS * 4 + 112 * 4)

typedef _Float16 v16h __attribute__((ext_vector_type(16)));
typedef float    v8f  __attribute__((ext_vector_type(8)));

// reduce within each 16-lane half of a wave32 (offsets 1..8 never cross halves)
__device__ __forceinline__ float red16(float v) {
#pragma unroll
  for (int off = 1; off < 16; off <<= 1) v += __shfl_xor(v, off, 32);
  return v;
}

__global__ __launch_bounds__(256)
void din_fused(const int* __restrict__ user_idx,
               const int* __restrict__ item_idx,
               const int* __restrict__ user_hist,
               const float* __restrict__ ug,          // user_embed_global [64]
               const float* __restrict__ tgt_w,       // item_embed_target_w [(N_ITEMS+1)x64]
               const float* __restrict__ hist_w,      // item_embed_hist_w   [(N_ITEMS+1)x64]
               const float* __restrict__ aff_user,    // [N_USERS x 128]
               const float* __restrict__ aff_item,    // [(N_ITEMS+1) x 128]
               const float* __restrict__ W1,          // [256 x 128]
               const float* __restrict__ b1,
               const float* __restrict__ g1,
               const float* __restrict__ be1,
               const float* __restrict__ W2,          // [128 x 64]
               const float* __restrict__ b2,
               const float* __restrict__ g2,
               const float* __restrict__ be2,
               const float* __restrict__ mW1,         // [128 x 64]
               const float* __restrict__ mb1,
               const float* __restrict__ mg1,
               const float* __restrict__ mbe1,
               const float* __restrict__ mW2,         // [64 x 32]
               const float* __restrict__ mb2,
               const float* __restrict__ mg2,
               const float* __restrict__ mbe2,
               const float* __restrict__ pW,          // [32]
               const float* __restrict__ pb,          // [1]
               float* __restrict__ out)               // [B]
{
  extern __shared__ unsigned char smem[];
  _Float16* sW1 = (_Float16*)smem;           // fragment-order f16 W1
  _Float16* sW2 = sW1 + SW1_H;               // fragment-order f16 W2
  _Float16* sUE = sW2 + SW2_H;               // user affection embedding (f16)
  float* sG1  = (float*)(sUE + SUE_H);
  float* sBe1 = sG1  + 128;
  float* sB1  = sBe1 + 128;
  float* sG2  = sB1  + 128;
  float* sBe2 = sG2  + 64;
  float* sB2  = sBe2 + 64;
  float* sUG  = sB2  + 64;
  float* sScore = sUG + 64;                  // 112 (scores -> softmax weights)
  float* sRed = sScore + 112;                // 8
  float* sVec = sRed + 8;                    // 256: cat[0..127], z1[128..191], z2[192..223]
  float* sPart = sVec + 256;                 // 256: context partials
  int* sIdx = (int*)(sPart + 256);           // 112 history item indices
  _Float16* sH1 = (_Float16*)(sIdx + 112);   // 7 waves x 16 x H1_LD f16 activations

  const int b    = blockIdx.x;
  const int tid  = threadIdx.x;
  const int lane = tid & 31;
  const int wave = tid >> 5;
  const int user = user_idx[b];
  const int item = item_idx[b];

  // ---------------- stage weights into WMMA B-fragment order ----------------
  // B fragment (32xK=32 step, N=16): lane = nn + 16*(kk>=16), half = kk&15
  for (int e = tid; e < 32768; e += 256) {
    int k = e >> 7, c = e & 127;
    int s = k >> 5, kk = k & 31, n = c >> 4, nn = c & 15;
    int ln = nn + ((kk & 16) ? 16 : 0);
    sW1[(s * 8 + n) * 512 + ln * 16 + (kk & 15)] = (_Float16)W1[e];
  }
  for (int e = tid; e < 8192; e += 256) {
    int k = e >> 6, c = e & 63;
    int s = k >> 5, kk = k & 31, n = c >> 4, nn = c & 15;
    int ln = nn + ((kk & 16) ? 16 : 0);
    sW2[(s * 4 + n) * 512 + ln * 16 + (kk & 15)] = (_Float16)W2[e];
  }
  if (tid < 128) {
    sUE[tid] = (_Float16)aff_user[(size_t)user * 128 + tid];
    sG1[tid] = g1[tid]; sBe1[tid] = be1[tid]; sB1[tid] = b1[tid];
  }
  if (tid < 64) {
    sG2[tid] = g2[tid]; sBe2[tid] = be2[tid]; sB2[tid] = b2[tid];
    sUG[tid] = ug[tid];
  }
  if (tid < HP) sIdx[tid] = (tid < HLEN) ? user_hist[(size_t)user * HLEN + tid] : N_ITEMS;
  __syncthreads();

  // ---------------- per-wave 16-row tile: affection MLP + scores ----------------
  if (wave < NTILE) {
    const int tile = wave;
    const int m  = lane & 15;
    const int kb = (lane & 16) ? 16 : 0;        // 16-bit A-matrix lane K offset
    const int rowoff = (lane >> 4) << 3;        // C-matrix: upper lanes hold rows +8
    const size_t refer = (size_t)sIdx[tile * 16 + m];
    const float* ie = aff_item + refer * 128;

    // ---- layer 1: [16 x 256] @ [256 x 128], 8 K-steps x 8 N-tiles ----
    v8f acc[8];
#pragma unroll
    for (int n = 0; n < 8; ++n)
#pragma unroll
      for (int j = 0; j < 8; ++j) acc[n][j] = 0.0f;

#pragma unroll
    for (int s = 0; s < 8; ++s) {
      v16h a;
      if (s < 4) {                               // x[k<128] = user embedding (LDS bcast)
        const int k0 = s * 32;
#pragma unroll
        for (int i = 0; i < 8; ++i) {
          a[i]     = sUE[k0 + kb + i];
          a[8 + i] = sUE[k0 + kb + 8 + i];
        }
      } else {                                   // x[k>=128] = gathered aff_item row
        const float* p = ie + (s - 4) * 32;
#pragma unroll
        for (int i = 0; i < 8; ++i) {
          a[i]     = (_Float16)p[kb + i];
          a[8 + i] = (_Float16)p[kb + 8 + i];
        }
      }
#pragma unroll
      for (int n = 0; n < 8; ++n) {
        const _Float16* bp = sW1 + (s * 8 + n) * 512 + lane * 16;
        v16h bf;
#pragma unroll
        for (int i = 0; i < 16; ++i) bf[i] = bp[i];
        acc[n] = __builtin_amdgcn_wmma_f32_16x16x32_f16(
            false, a, false, bf, (short)0, acc[n], false, false);
      }
    }

    // ---- bias + LayerNorm + ReLU, fully in registers; f16 result -> LDS for layer 2 A ----
    _Float16* h1 = sH1 + wave * 16 * H1_LD;
#pragma unroll
    for (int v = 0; v < 8; ++v) {
      float sum = 0.f, sq = 0.f;
#pragma unroll
      for (int n = 0; n < 8; ++n) {
        const int col = n * 16 + (lane & 15);
        const float x = acc[n][v] + sB1[col];
        sum += x; sq += x * x;
      }
      sum = red16(sum); sq = red16(sq);
      const float mu  = sum * (1.0f / 128.0f);
      const float var = sq * (1.0f / 128.0f) - mu * mu;
      const float rs  = rsqrtf(var + LN_EPS);
      const int row = v + rowoff;
#pragma unroll
      for (int n = 0; n < 8; ++n) {
        const int col = n * 16 + (lane & 15);
        const float x = acc[n][v] + sB1[col];
        float y = (x - mu) * rs * sG1[col] + sBe1[col];
        h1[row * H1_LD + col] = (_Float16)fmaxf(y, 0.0f);
      }
    }

    // ---- layer 2: [16 x 128] @ [128 x 64], 4 K-steps x 4 N-tiles ----
    v8f acc2[4];
#pragma unroll
    for (int n = 0; n < 4; ++n)
#pragma unroll
      for (int j = 0; j < 8; ++j) acc2[n][j] = 0.0f;

#pragma unroll
    for (int s = 0; s < 4; ++s) {
      v16h a;
      const _Float16* hr = h1 + m * H1_LD + s * 32;
#pragma unroll
      for (int i = 0; i < 8; ++i) {
        a[i]     = hr[kb + i];
        a[8 + i] = hr[kb + 8 + i];
      }
#pragma unroll
      for (int n = 0; n < 4; ++n) {
        const _Float16* bp = sW2 + (s * 4 + n) * 512 + lane * 16;
        v16h bf;
#pragma unroll
        for (int i = 0; i < 16; ++i) bf[i] = bp[i];
        acc2[n] = __builtin_amdgcn_wmma_f32_16x16x32_f16(
            false, a, false, bf, (short)0, acc2[n], false, false);
      }
    }

    // ---- bias + LN + ReLU + dot(user_embed_global) fused, no LDS round-trip ----
#pragma unroll
    for (int v = 0; v < 8; ++v) {
      float sum = 0.f, sq = 0.f;
#pragma unroll
      for (int n = 0; n < 4; ++n) {
        const int col = n * 16 + (lane & 15);
        const float x = acc2[n][v] + sB2[col];
        sum += x; sq += x * x;
      }
      sum = red16(sum); sq = red16(sq);
      const float mu  = sum * (1.0f / 64.0f);
      const float var = sq * (1.0f / 64.0f) - mu * mu;
      const float rs  = rsqrtf(var + LN_EPS);
      float sc = 0.f;
#pragma unroll
      for (int n = 0; n < 4; ++n) {
        const int col = n * 16 + (lane & 15);
        const float x = acc2[n][v] + sB2[col];
        float y = fmaxf((x - mu) * rs * sG2[col] + sBe2[col], 0.0f);
        sc += y * sUG[col];
      }
      sc = red16(sc);
      const int row = tile * 16 + v + rowoff;
      if ((lane & 15) == 0) {
        const int ri = sIdx[row];
        const bool valid = (row < HLEN) && (ri != item) && (ri != N_ITEMS);
        sScore[row] = valid ? sc * 0.125f : -3.0e38f;   // 1/sqrt(64)
      }
    }
  }
  __syncthreads();

  // ---------------- softmax over 112 scores (wave 0) ----------------
  if (wave == 0) {
    float vs[4]; float vmax = -3.4e38f;
#pragma unroll
    for (int j = 0; j < 4; ++j) {
      int idx = lane + j * 32;
      vs[j] = (idx < HP) ? sScore[idx] : -3.4e38f;
      vmax = fmaxf(vmax, vs[j]);
    }
#pragma unroll
    for (int off = 1; off < 32; off <<= 1) vmax = fmaxf(vmax, __shfl_xor(vmax, off, 32));
    float ssum = 0.f;
#pragma unroll
    for (int j = 0; j < 4; ++j) {
      int idx = lane + j * 32;
      if (idx < HP) { float e = __expf(vs[j] - vmax); sScore[idx] = e; ssum += e; }
    }
#pragma unroll
    for (int off = 1; off < 32; off <<= 1) ssum += __shfl_xor(ssum, off, 32);
    const float inv = 1.0f / ssum;
#pragma unroll
    for (int j = 0; j < 4; ++j) {
      int idx = lane + j * 32;
      if (idx < HP) sScore[idx] *= inv;
    }
  }
  __syncthreads();

  // ---------------- weighted pooling of V over history (all 256 threads) ----------------
  {
    const int d = tid & 63, g = tid >> 6;      // 4 history groups of 25
    float p = 0.f;
    for (int h = g * 25; h < g * 25 + 25; ++h)
      p += sScore[h] * hist_w[(size_t)sIdx[h] * 64 + d];
    sPart[tid] = p;
  }
  __syncthreads();
  if (tid < 64) {
    sVec[tid] = sPart[tid] + sPart[64 + tid] + sPart[128 + tid] + sPart[192 + tid];
    sVec[64 + tid] = tgt_w[(size_t)item * 64 + tid];
  }
  __syncthreads();

  // ---------------- matcher MLP: 128 -> 64 -> 32 -> 1 (tiny; plain VALU) ----------------
  if (tid < 64) {
    float s = mb1[tid];
    for (int k = 0; k < 128; ++k) s += sVec[k] * mW1[k * 64 + tid];
    sVec[128 + tid] = s;
  }
  __syncthreads();
  if (tid == 0) {
    float sum = 0.f, sq = 0.f;
    for (int k = 0; k < 64; ++k) { float x = sVec[128 + k]; sum += x; sq += x * x; }
    float mu = sum * (1.0f / 64.0f);
    sRed[0] = mu; sRed[1] = rsqrtf(sq * (1.0f / 64.0f) - mu * mu + LN_EPS);
  }
  __syncthreads();
  if (tid < 64) {
    float y = (sVec[128 + tid] - sRed[0]) * sRed[1] * mg1[tid] + mbe1[tid];
    sVec[128 + tid] = fmaxf(y, 0.f);
  }
  __syncthreads();
  if (tid < 32) {
    float s = mb2[tid];
    for (int k = 0; k < 64; ++k) s += sVec[128 + k] * mW2[k * 32 + tid];
    sVec[192 + tid] = s;
  }
  __syncthreads();
  if (tid == 0) {
    float sum = 0.f, sq = 0.f;
    for (int k = 0; k < 32; ++k) { float x = sVec[192 + k]; sum += x; sq += x * x; }
    float mu = sum * (1.0f / 32.0f);
    sRed[0] = mu; sRed[1] = rsqrtf(sq * (1.0f / 32.0f) - mu * mu + LN_EPS);
  }
  __syncthreads();
  if (tid < 32) {
    float y = (sVec[192 + tid] - sRed[0]) * sRed[1] * mg2[tid] + mbe2[tid];
    sVec[192 + tid] = fmaxf(y, 0.f);
  }
  __syncthreads();
  if (tid == 0) {
    float lg = pb[0];
    for (int k = 0; k < 32; ++k) lg += sVec[192 + k] * pW[k];
    out[b] = lg;
  }
}

extern "C" void kernel_launch(void* const* d_in, const int* in_sizes, int n_in,
                              void* d_out, int out_size, void* d_ws, size_t ws_size,
                              hipStream_t stream) {
  (void)in_sizes; (void)n_in; (void)out_size; (void)d_ws; (void)ws_size;
  const int*   user_idx = (const int*)d_in[0];
  const int*   item_idx = (const int*)d_in[1];
  const int*   user_hist = (const int*)d_in[2];
  const float* ug    = (const float*)d_in[3];
  const float* tgt_w = (const float*)d_in[4];
  const float* hist_w = (const float*)d_in[5];
  const float* aff_user = (const float*)d_in[6];
  const float* aff_item = (const float*)d_in[7];
  const float* W1 = (const float*)d_in[8];
  const float* b1 = (const float*)d_in[9];
  const float* g1 = (const float*)d_in[10];
  const float* be1 = (const float*)d_in[11];
  const float* W2 = (const float*)d_in[12];
  const float* b2 = (const float*)d_in[13];
  const float* g2 = (const float*)d_in[14];
  const float* be2 = (const float*)d_in[15];
  const float* mW1 = (const float*)d_in[16];
  const float* mb1 = (const float*)d_in[17];
  const float* mg1 = (const float*)d_in[18];
  const float* mbe1 = (const float*)d_in[19];
  const float* mW2 = (const float*)d_in[20];
  const float* mb2 = (const float*)d_in[21];
  const float* mg2 = (const float*)d_in[22];
  const float* mbe2 = (const float*)d_in[23];
  const float* pW = (const float*)d_in[24];
  const float* pb = (const float*)d_in[25];
  float* out = (float*)d_out;

  // allow > default dynamic LDS (idempotent, host-side; safe under graph capture)
  (void)hipFuncSetAttribute(reinterpret_cast<const void*>(&din_fused),
                            hipFuncAttributeMaxDynamicSharedMemorySize,
                            (int)SMEM_BYTES);

  din_fused<<<BATCH, 256, SMEM_BYTES, stream>>>(
      user_idx, item_idx, user_hist, ug, tgt_w, hist_w, aff_user, aff_item,
      W1, b1, g1, be1, W2, b2, g2, be2,
      mW1, mb1, mg1, mbe1, mW2, mb2, mg2, mbe2, pW, pb, out);
}
